// TidalTransformer_71897752535312
// MI455X (gfx1250) — matrix operand order repaired and checked by reading, not compiled, not run on hardware
//
#include <hip/hip_runtime.h>
#include <hip/hip_bf16.h>

#define BB  4
#define SS  1024
#define HH  1024
#define NH_ 16
#define DK_ 64
#define DF_ 4096
#define LL  4
#define VV  32000
#define MM  (BB * SS)

typedef __attribute__((ext_vector_type(4)))  float  v4f;
typedef __attribute__((ext_vector_type(8)))  float  v8f;
typedef __attribute__((ext_vector_type(8)))  __bf16 v8bf;
typedef __attribute__((ext_vector_type(16))) __bf16 v16bf;

// ---------------------------------------------------------------- helpers

__device__ __forceinline__ v8f zero8() {
  v8f z;
#pragma unroll
  for (int i = 0; i < 8; ++i) z[i] = 0.0f;
  return z;
}

__device__ __forceinline__ v8f wmma_bf16(v16bf a, v16bf b, v8f c) {
  // D = A(16x32 bf16) * B(32x16 bf16) + C(16x16 f32)
  return __builtin_amdgcn_wmma_f32_16x16x32_bf16(false, a, false, b,
                                                 (short)0, c, false, false);
}

__device__ __forceinline__ v16bf cvt16(v4f a0, v4f a1, v4f a2, v4f a3) {
  v16bf r;
#pragma unroll
  for (int i = 0; i < 4; ++i) {
    r[i]      = (__bf16)a0[i];
    r[i + 4]  = (__bf16)a1[i];
    r[i + 8]  = (__bf16)a2[i];
    r[i + 12] = (__bf16)a3[i];
  }
  return r;
}

// A-matrix 16x32 bf16 fragment from an fp32 row (per-lane K runs:
// lanes 0-15 -> K {k0+0..7, k0+16..23}; lanes 16-31 -> K {k0+8..15, k0+24..31})
__device__ __forceinline__ v16bf load_a_f32(const float* row, int k0, int half) {
  const float* p0 = row + k0 + 8 * half;
  const float* p1 = row + k0 + 16 + 8 * half;
  return cvt16(*(const v4f*)p0, *(const v4f*)(p0 + 4),
               *(const v4f*)p1, *(const v4f*)(p1 + 4));
}

__device__ __forceinline__ v16bf cat8(v8bf lo, v8bf hi) {
  return __builtin_shufflevector(lo, hi, 0, 1, 2, 3, 4, 5, 6, 7,
                                 8, 9, 10, 11, 12, 13, 14, 15);
}

// A fragment straight from bf16 memory (global or LDS)
__device__ __forceinline__ v16bf load_a_bf(const __bf16* row, int k0, int half) {
  return cat8(*(const v8bf*)(row + k0 + 8 * half),
              *(const v8bf*)(row + k0 + 16 + 8 * half));
}

// B fragment straight from bf16 memory: row of B^T (N x K)
// lanes 0-15 -> K {k0+0..7, k0+16..23} ; lanes 16-31 -> K {k0+8..15, k0+24..31}
__device__ __forceinline__ v16bf load_b_bf(const __bf16* row, int k0, int half) {
  const __bf16* p = row + k0 + 8 * half;
  return cat8(*(const v8bf*)p, *(const v8bf*)(p + 16));
}

// ---------------------------------------------------------------- embed

__global__ __launch_bounds__(256)
void embed_kernel(const int* __restrict__ ids, const float* __restrict__ emb,
                  float* __restrict__ x) {
  const int row = blockIdx.x;
  const int id  = ids[row];
  const v4f* src = (const v4f*)(emb + (size_t)id * HH);
  v4f* dst = (v4f*)(x + (size_t)row * HH);
  v4f v = src[threadIdx.x];
#pragma unroll
  for (int i = 0; i < 4; ++i) v[i] *= 32.0f;  // sqrt(H) = 32
  dst[threadIdx.x] = v;
}

// ---------------------------------------------------------------- GEMM
// C = act(A[M,K] @ W[K,N] + bias). Block tile 128x128, 8 waves of 32x64,
// 8 WMMA per 32-wide k-step. W tile staged to LDS transposed as bf16.
// out_mode: 0 = fp32 row-major (M,N); 1 = bf16 (b,h,s,d); 2 = bf16 (b,h,d,s).

__global__ __launch_bounds__(256)
void gemm_wmma_kernel(const float* __restrict__ A, const float* __restrict__ W,
                      const float* __restrict__ bias, float* __restrict__ C,
                      int M, int N, int K, int out_mode, int do_gelu) {
  (void)M;
  __shared__ __attribute__((aligned(16))) __bf16 wt[128][40];  // [n][k], 80B stride

  const int n0 = blockIdx.x * 128;
  const int m0 = blockIdx.y * 128;
  const int t    = threadIdx.x;
  const int lane = t & 31;
  const int wave = t >> 5;
  const int half = lane >> 4;
  const int ln   = lane & 15;
  const int m_base = (wave >> 1) * 32;   // 0,32,64,96
  const int n_base = (wave & 1) * 64;    // 0,64

  const float* arow0 = A + (size_t)(m0 + m_base + ln) * K;
  const float* arow1 = A + (size_t)(m0 + m_base + 16 + ln) * K;

  v8f acc[2][4];
#pragma unroll
  for (int i = 0; i < 2; ++i)
#pragma unroll
    for (int j = 0; j < 4; ++j) acc[i][j] = zero8();

  const int kr = t >> 3;          // 0..31 (k row in tile)
  const int nq = (t & 7) * 16;    // 0..112 (n column group)

  for (int k0 = 0; k0 < K; k0 += 32) {
    // stage W[k0:k0+32, n0:n0+128] -> LDS transposed (bf16)
    const float* wp = W + (size_t)(k0 + kr) * N + n0 + nq;
    v4f w0 = *(const v4f*)wp;
    v4f w1 = *(const v4f*)(wp + 4);
    v4f w2 = *(const v4f*)(wp + 8);
    v4f w3 = *(const v4f*)(wp + 12);
    if (k0 + 32 < K) __builtin_prefetch(wp + (size_t)32 * N, 0, 1);
#pragma unroll
    for (int i = 0; i < 4; ++i) {
      wt[nq + i][kr]      = (__bf16)w0[i];
      wt[nq + 4 + i][kr]  = (__bf16)w1[i];
      wt[nq + 8 + i][kr]  = (__bf16)w2[i];
      wt[nq + 12 + i][kr] = (__bf16)w3[i];
    }
    __syncthreads();

    const v16bf a0 = load_a_f32(arow0, k0, half);
    const v16bf a1 = load_a_f32(arow1, k0, half);
#pragma unroll
    for (int fn = 0; fn < 4; ++fn) {
      const v16bf bf = load_b_bf(&wt[n_base + fn * 16 + ln][0], 0, half);
      acc[0][fn] = wmma_bf16(a0, bf, acc[0][fn]);
      acc[1][fn] = wmma_bf16(a1, bf, acc[1][fn]);
    }
    __syncthreads();
  }

  // epilogue: C layout -> VGPR r: row m = frag_m*16 + 8*half + r, col = ln
  __bf16* Cb = (__bf16*)C;
#pragma unroll
  for (int fm = 0; fm < 2; ++fm)
#pragma unroll
    for (int fn = 0; fn < 4; ++fn) {
      const int ng = n0 + n_base + fn * 16 + ln;
      const float bv = bias ? bias[ng] : 0.0f;
#pragma unroll
      for (int r = 0; r < 8; ++r) {
        const int mg = m0 + m_base + fm * 16 + half * 8 + r;
        float v = acc[fm][fn][r] + bv;
        if (do_gelu) v = 0.5f * v * (1.0f + erff(v * 0.70710678118f));
        if (out_mode == 0) {
          C[(size_t)mg * N + ng] = v;
        } else {
          const int b = mg >> 10, s = mg & 1023;   // M = B*S, S = 1024
          const int h = ng >> 6,  d = ng & 63;     // N = H, DK = 64
          if (out_mode == 1)
            Cb[(((size_t)(b * NH_ + h) * SS) + s) * DK_ + d] = (__bf16)v;  // (b,h,s,d)
          else
            Cb[(((size_t)(b * NH_ + h) * DK_) + d) * SS + s] = (__bf16)v;  // (b,h,d,s)
        }
      }
    }
}

// ---------------------------------------------------------------- attention
// Flash-style per (b,h) and q-block of 64. 4 waves, each owns 16 q rows.
// Q, K, V^T are bf16 -> WMMA operands load directly, zero conversions.

__global__ __launch_bounds__(128)
void attn_kernel(const __bf16* __restrict__ qb, const __bf16* __restrict__ kbuf,
                 const __bf16* __restrict__ vtb, const int* __restrict__ start_pos,
                 float* __restrict__ outb) {
  __shared__ __attribute__((aligned(16))) __bf16 lds_p[4][16][72];  // per-wave P tile

  const int bh = blockIdx.y;
  const int b = bh >> 4, h = bh & 15;
  const int sp = start_pos[b];
  const float slope = exp2f(-0.5f * (float)(h + 1));  // 2^(-8*(h+1)/NH)

  const int wave = threadIdx.x >> 5;
  const int lane = threadIdx.x & 31;
  const int half = lane >> 4, ln = lane & 15;
  const int qbase = blockIdx.x * 64 + wave * 16;

  const __bf16* qrow = qb + ((size_t)bh * SS + qbase + ln) * DK_;
  const v16bf aq0 = load_a_bf(qrow, 0, half);
  const v16bf aq1 = load_a_bf(qrow, 32, half);

  float mstate[8], lstate[8];
  v8f o[4];
#pragma unroll
  for (int r = 0; r < 8; ++r) { mstate[r] = -3.0e38f; lstate[r] = 0.0f; }
#pragma unroll
  for (int fd = 0; fd < 4; ++fd) o[fd] = zero8();

  for (int kblk = 0; kblk < SS / 64; ++kblk) {
    // ---- scores S = Q K^T / sqrt(DK) + alibi, masked
    v8f sf[4];
#pragma unroll
    for (int fn = 0; fn < 4; ++fn) {
      const __bf16* krow = kbuf + ((size_t)bh * SS + kblk * 64 + fn * 16 + ln) * DK_;
      v8f s = zero8();
      s = wmma_bf16(aq0, load_b_bf(krow, 0, half), s);
      s = wmma_bf16(aq1, load_b_bf(krow, 32, half), s);
      sf[fn] = s;
    }
#pragma unroll
    for (int fn = 0; fn < 4; ++fn) {
      const int kg = kblk * 64 + fn * 16 + ln;
#pragma unroll
      for (int r = 0; r < 8; ++r) {
        const int qg = qbase + half * 8 + r;
        const float sc = sf[fn][r] * 0.125f + slope * (float)(kg - sp);
        const bool keep = (kg <= qg) | (qg < sp) | (kg < sp);
        sf[fn][r] = keep ? sc : -1.0e9f;
      }
    }
    // ---- online softmax (row reductions across 16-lane half-groups)
#pragma unroll
    for (int r = 0; r < 8; ++r) {
      float mb = fmaxf(fmaxf(sf[0][r], sf[1][r]), fmaxf(sf[2][r], sf[3][r]));
      mb = fmaxf(mb, __shfl_xor(mb, 1, 16));
      mb = fmaxf(mb, __shfl_xor(mb, 2, 16));
      mb = fmaxf(mb, __shfl_xor(mb, 4, 16));
      mb = fmaxf(mb, __shfl_xor(mb, 8, 16));
      const float mn = fmaxf(mstate[r], mb);
      const float sc = __expf(mstate[r] - mn);
      mstate[r] = mn;
      float rs = 0.0f;
#pragma unroll
      for (int fn = 0; fn < 4; ++fn) {
        const float e = __expf(sf[fn][r] - mn);
        sf[fn][r] = e;
        rs += e;
      }
      rs += __shfl_xor(rs, 1, 16);
      rs += __shfl_xor(rs, 2, 16);
      rs += __shfl_xor(rs, 4, 16);
      rs += __shfl_xor(rs, 8, 16);
      lstate[r] = lstate[r] * sc + rs;
#pragma unroll
      for (int fd = 0; fd < 4; ++fd) o[fd][r] *= sc;
    }
    // ---- relayout P (C layout -> A layout) through LDS
#pragma unroll
    for (int fn = 0; fn < 4; ++fn)
#pragma unroll
      for (int r = 0; r < 8; ++r)
        lds_p[wave][half * 8 + r][fn * 16 + ln] = (__bf16)sf[fn][r];
    __syncthreads();
    const __bf16* prow = &lds_p[wave][ln][0];
    const v16bf pa0 = load_a_bf(prow, 0, half);
    const v16bf pa1 = load_a_bf(prow, 32, half);
    // ---- O += P @ V  (V^T rows are contiguous over keys, bf16)
#pragma unroll
    for (int fd = 0; fd < 4; ++fd) {
      const __bf16* vrow = vtb + ((size_t)bh * DK_ + fd * 16 + ln) * SS + kblk * 64;
      o[fd] = wmma_bf16(pa0, load_b_bf(vrow, 0, half), o[fd]);
      o[fd] = wmma_bf16(pa1, load_b_bf(vrow, 32, half), o[fd]);
    }
    __syncthreads();
  }
  // ---- normalize and store to (B, S, H) so the Wo GEMM reads row-major
#pragma unroll
  for (int fd = 0; fd < 4; ++fd)
#pragma unroll
    for (int r = 0; r < 8; ++r) {
      const int qg = qbase + half * 8 + r;
      const float v = o[fd][r] / lstate[r];
      outb[((size_t)(b * SS + qg) * HH) + h * DK_ + fd * 16 + ln] = v;
    }
}

// ---------------------------------------------------------------- residual+LN

__global__ __launch_bounds__(256)
void add_ln_kernel(float* __restrict__ x, const float* __restrict__ tadd,
                   const float* __restrict__ g, const float* __restrict__ bta) {
  __shared__ float red[2][8];
  const size_t base = (size_t)blockIdx.x * HH;
  const int t = threadIdx.x;
  float vals[4];
  float s = 0.0f, s2 = 0.0f;
#pragma unroll
  for (int j = 0; j < 4; ++j) {
    const int i = t + j * 256;
    const float v = x[base + i] + tadd[base + i];
    vals[j] = v;
    s += v;
    s2 += v * v;
  }
#pragma unroll
  for (int off = 16; off >= 1; off >>= 1) {
    s  += __shfl_xor(s, off, 32);
    s2 += __shfl_xor(s2, off, 32);
  }
  if ((t & 31) == 0) { red[0][t >> 5] = s; red[1][t >> 5] = s2; }
  __syncthreads();
  float ts = 0.0f, ts2 = 0.0f;
#pragma unroll
  for (int w = 0; w < 8; ++w) { ts += red[0][w]; ts2 += red[1][w]; }
  const float mu  = ts * (1.0f / HH);
  const float var = ts2 * (1.0f / HH) - mu * mu;
  const float rs  = rsqrtf(var + 1e-5f);
#pragma unroll
  for (int j = 0; j < 4; ++j) {
    const int i = t + j * 256;
    x[base + i] = (vals[j] - mu) * rs * g[i] + bta[i];
  }
}

// ---------------------------------------------------------------- output mask

__global__ __launch_bounds__(256)
void mask_kernel(float* __restrict__ logits, const int* __restrict__ start_pos) {
  const int row = blockIdx.x;
  const int b = row >> 10, s = row & 1023;
  if (s >= start_pos[b]) return;
  float* p = logits + (size_t)row * VV;
  for (int i = threadIdx.x; i < VV; i += 256) p[i] = 0.0f;
}

// ---------------------------------------------------------------- driver

extern "C" void kernel_launch(void* const* d_in, const int* in_sizes, int n_in,
                              void* d_out, int out_size, void* d_ws, size_t ws_size,
                              hipStream_t stream) {
  (void)in_sizes; (void)n_in; (void)out_size; (void)ws_size;
  const int*   ids  = (const int*)d_in[0];
  const int*   spos = (const int*)d_in[1];
  const float* emb  = (const float*)d_in[2];
  const float* Wq   = (const float*)d_in[3];
  const float* bq   = (const float*)d_in[4];
  const float* Wk   = (const float*)d_in[5];
  const float* bk   = (const float*)d_in[6];
  const float* Wv   = (const float*)d_in[7];
  const float* bv   = (const float*)d_in[8];
  const float* Wo   = (const float*)d_in[9];
  const float* bo   = (const float*)d_in[10];
  const float* ln1g = (const float*)d_in[11];
  const float* ln1b = (const float*)d_in[12];
  const float* ln2g = (const float*)d_in[13];
  const float* ln2b = (const float*)d_in[14];
  const float* W1   = (const float*)d_in[15];
  const float* b1   = (const float*)d_in[16];
  const float* W2   = (const float*)d_in[17];
  const float* b2   = (const float*)d_in[18];
  const float* Wf   = (const float*)d_in[19];
  const float* bf_  = (const float*)d_in[20];

  float* out = (float*)d_out;
  const size_t T = (size_t)MM * HH;  // 4,194,304 elements

  // x / tmp live in d_ws (32 MB). Attention scratch (bf16 Q/K/V^T) and FFN
  // scratch live in the front of d_out (<= 8*T floats << 131M floats);
  // the final vocab GEMM rewrites all of d_out, keeping replays deterministic.
  float*  x   = (float*)d_ws;
  float*  tmp = x + T;
  __bf16* qb  = (__bf16*)out;              // T bf16
  __bf16* kb  = (__bf16*)(out + T);        // T bf16
  __bf16* vt  = (__bf16*)(out + 2 * T);    // T bf16
  float*  at  = out + 3 * T;               // T fp32
  float*  ff  = out + 4 * T;               // 4*T fp32 (M x DF)

  const dim3 blk256(256), blk128(128);
  const dim3 gH(HH / 128, MM / 128);

  embed_kernel<<<MM, blk256, 0, stream>>>(ids, emb, x);

  for (int l = 0; l < LL; ++l) {
    const size_t wofs = (size_t)l * HH * HH;
    gemm_wmma_kernel<<<gH, blk256, 0, stream>>>(x, Wq + wofs, bq + l * HH, (float*)qb, MM, HH, HH, 1, 0);
    gemm_wmma_kernel<<<gH, blk256, 0, stream>>>(x, Wk + wofs, bk + l * HH, (float*)kb, MM, HH, HH, 1, 0);
    gemm_wmma_kernel<<<gH, blk256, 0, stream>>>(x, Wv + wofs, bv + l * HH, (float*)vt, MM, HH, HH, 2, 0);
    attn_kernel<<<dim3(SS / 64, BB * NH_), blk128, 0, stream>>>(qb, kb, vt, spos, at);
    gemm_wmma_kernel<<<gH, blk256, 0, stream>>>(at, Wo + wofs, bo + l * HH, tmp, MM, HH, HH, 0, 0);
    add_ln_kernel<<<MM, blk256, 0, stream>>>(x, tmp, ln1g + l * HH, ln1b + l * HH);
    gemm_wmma_kernel<<<dim3(DF_ / 128, MM / 128), blk256, 0, stream>>>(
        x, W1 + (size_t)l * HH * DF_, b1 + l * DF_, ff, MM, DF_, HH, 0, 1);
    gemm_wmma_kernel<<<gH, blk256, 0, stream>>>(
        ff, W2 + (size_t)l * DF_ * HH, b2 + l * HH, tmp, MM, HH, DF_, 0, 0);
    add_ln_kernel<<<MM, blk256, 0, stream>>>(x, tmp, ln2g + l * HH, ln2b + l * HH);
  }

  gemm_wmma_kernel<<<dim3(VV / 128, MM / 128), blk256, 0, stream>>>(
      x, Wf, bf_, out, MM, VV, HH, 0, 0);
  mask_kernel<<<MM, blk256, 0, stream>>>(out, spos);
}
